// SAGPool_ROI_18657337933846
// MI455X (gfx1250) — matrix operand order repaired and checked by reading, not compiled, not run on hardware
//
#include <hip/hip_runtime.h>
#include <hip/hip_fp16.h>

typedef __attribute__((ext_vector_type(16))) _Float16 v16h;
typedef __attribute__((ext_vector_type(8)))  float    v8f;

#define GG 1024
#define RR 273
#define KK 219            // ceil(0.8 * 273)
#define CC 256
#define DEG 32
#define NN (GG * RR)      // 279552
#define EE (GG * RR * DEG) // 8945664

// One-time convert of W [C] f32 -> f16 table in workspace (512 bytes, L2-hot).
__global__ __launch_bounds__(256)
void convert_w(const float* __restrict__ W, _Float16* __restrict__ W16) {
    int i = threadIdx.x;            // single block, blockDim = CC
    W16[i] = (_Float16)W[i];
}

// ---------------------------------------------------------------------------
// Score matvec h = x @ W^T via V_WMMA_F32_16X16X32_F16.
// One wave handles 16 node rows; 8 WMMAs accumulate over C=256.
// A k order used here (contiguous, not the ISA table order): lane<16 -> M=lane, K in {0..15};
//                          lane>=16 -> M=lane-16, K in {16..31}.
// B: W replicated in ALL 16 columns (lane<16 -> K 0..15, lane>=16 -> K 16..31),
//    so every column of D is the score vector — no masking, layout-robust.
// D: lane 0 holds rows 0..7 in c[0..7]; lane 16 holds rows 8..15.
// ---------------------------------------------------------------------------
__global__ __launch_bounds__(256)
void score_matvec_wmma(const float* __restrict__ x,
                       const _Float16* __restrict__ W16,
                       float* __restrict__ h) {
    const int lane      = threadIdx.x & 31;
    const int wave      = threadIdx.x >> 5;
    const int node_base = blockIdx.x * 128 + wave * 16;   // 8 waves * 16 nodes
    const int m         = lane & 15;
    const int node      = node_base + m;
    const int khalf_a   = (lane < 16) ? 0 : 16;
    const int khalf_b   = (lane < 16) ? 0 : 16;

    const float* __restrict__ xrow = x + (size_t)node * CC;
    v8f c = {};

    #pragma unroll
    for (int kc = 0; kc < CC; kc += 32) {
        v16h a;
        #pragma unroll
        for (int i = 0; i < 8; ++i) {
            a[i]     = (_Float16)xrow[kc + khalf_a + i];
            a[8 + i] = (_Float16)xrow[kc + khalf_a + 8 + i];
        }
        // 32-byte aligned vector load of 16 f16 W values (two b128 loads)
        v16h b = *(const v16h*)(W16 + kc + khalf_b);
        c = __builtin_amdgcn_wmma_f32_16x16x32_f16(
                /*neg_a=*/false, a, /*neg_b=*/false, b,
                /*c_mod=*/(short)0, c, /*reuse_a=*/false, /*reuse_b=*/false);
    }

    if (lane == 0) {
        #pragma unroll
        for (int i = 0; i < 8; ++i) h[node_base + i] = c[i];
    } else if (lane == 16) {
        #pragma unroll
        for (int i = 0; i < 8; ++i) h[node_base + 8 + i] = c[i];
    }
}

// deg starts at 1.0 (self-loop)
__global__ __launch_bounds__(256)
void init_deg(float* __restrict__ deg) {
    int n = blockIdx.x * 256 + threadIdx.x;
    if (n < NN) deg[n] = 1.0f;
}

__global__ __launch_bounds__(256)
void accum_deg(const int* __restrict__ dst, float* __restrict__ deg) {
    int e = blockIdx.x * 256 + threadIdx.x;
    if (e < EE) atomicAdd(&deg[dst[e]], 1.0f);
}

// dis = rsqrt(deg); acc = dis^2 * h  (self-loop term, non-atomic init)
__global__ __launch_bounds__(256)
void dis_self(const float* __restrict__ h, float* __restrict__ deg_dis,
              float* __restrict__ acc) {
    int n = blockIdx.x * 256 + threadIdx.x;
    if (n < NN) {
        float d = rsqrtf(deg_dis[n]);
        deg_dis[n] = d;
        acc[n] = d * d * h[n];
    }
}

__global__ __launch_bounds__(256)
void agg_edges(const int* __restrict__ src, const int* __restrict__ dst,
               const float* __restrict__ dis, const float* __restrict__ h,
               float* __restrict__ acc) {
    int e = blockIdx.x * 256 + threadIdx.x;
    if (e < EE) {
        int s = src[e], d = dst[e];
        atomicAdd(&acc[d], dis[s] * dis[d] * h[s]);
    }
}

__global__ __launch_bounds__(256)
void score_final(const float* __restrict__ acc, const float* __restrict__ b,
                 float* __restrict__ score_out) {
    int n = blockIdx.x * 256 + threadIdx.x;
    if (n < NN) score_out[n] = acc[n] + b[0];
}

// Per-graph stable top-K by rank counting (matches lax.top_k tie behavior).
__global__ __launch_bounds__(256)
void topk_kernel(const float* __restrict__ score, int* __restrict__ perm,
                 int* __restrict__ new_id, float* __restrict__ perm_out_f,
                 float* __restrict__ batch_new_f) {
    __shared__ float s[RR];
    const int g = blockIdx.x;
    const float* __restrict__ sg = score + g * RR;
    for (int i = threadIdx.x; i < RR; i += 256) s[i] = sg[i];
    __syncthreads();
    for (int i = threadIdx.x; i < RR; i += 256) {
        float si = s[i];
        int rank = 0;
        for (int j = 0; j < RR; ++j) {
            float sj = s[j];
            rank += (sj > si) || (sj == si && j < i);
        }
        int nid = -1;
        if (rank < KK) {
            int pos = g * KK + rank;
            perm[pos]        = g * RR + i;
            perm_out_f[pos]  = (float)(g * RR + i);
            batch_new_f[pos] = (float)g;
            nid = pos;
        }
        new_id[g * RR + i] = nid;
    }
}

// x_new[j,:] = x[perm[j],:] * tanh(score[perm[j]]) — float4 coalesced
__global__ __launch_bounds__(64)
void gather_scale(const float* __restrict__ x, const float* __restrict__ score,
                  const int* __restrict__ perm, float* __restrict__ x_new) {
    int j = blockIdx.x;            // 0 .. G*K-1
    int node = perm[j];
    float t = tanhf(score[node]);
    const float4* __restrict__ xr = (const float4*)(x + (size_t)node * CC);
    float4* __restrict__ orow = (float4*)(x_new + (size_t)j * CC);
    float4 v = xr[threadIdx.x];    // 64 threads * 16B = 1KB row
    float4 o; o.x = v.x * t; o.y = v.y * t; o.z = v.z * t; o.w = v.w * t;
    orow[threadIdx.x] = o;
}

__global__ __launch_bounds__(256)
void remap_edges(const int* __restrict__ src, const int* __restrict__ dst,
                 const int* __restrict__ new_id, float* __restrict__ ei_new,
                 float* __restrict__ keep_out) {
    int e = blockIdx.x * 256 + threadIdx.x;
    if (e < EE) {
        int ns = new_id[src[e]];
        int nd = new_id[dst[e]];
        bool keep = (ns >= 0) && (nd >= 0);
        ei_new[e]              = keep ? (float)ns : -1.0f;
        ei_new[(size_t)EE + e] = keep ? (float)nd : -1.0f;
        keep_out[e]            = keep ? 1.0f : 0.0f;
    }
}

extern "C" void kernel_launch(void* const* d_in, const int* in_sizes, int n_in,
                              void* d_out, int out_size, void* d_ws, size_t ws_size,
                              hipStream_t stream) {
    const float* x    = (const float*)d_in[0];
    const int*   edge = (const int*)d_in[1];
    // d_in[2] = batch (recomputed analytically), d_in[3] = W, d_in[4] = b
    const float* W = (const float*)d_in[3];
    const float* b = (const float*)d_in[4];

    const int* src = edge;
    const int* dst = edge + EE;

    // workspace layout (all offsets 32B-aligned)
    float*    h    = (float*)d_ws;          // [N]
    float*    dis  = h + NN;                // [N] (doubles as deg)
    float*    acc  = dis + NN;              // [N]
    int*      nid  = (int*)(acc + NN);      // [N]
    int*      perm = nid + NN;              // [G*K]
    _Float16* W16  = (_Float16*)(perm + (size_t)GG * KK);  // [C]

    // output layout (concatenated, return order)
    float* out      = (float*)d_out;
    float* o_xnew   = out;                                   // [G*K*C]
    float* o_ei     = o_xnew + (size_t)GG * KK * CC;         // [2*E]
    float* o_keep   = o_ei + 2 * (size_t)EE;                 // [E]
    float* o_batch  = o_keep + (size_t)EE;                   // [G*K]
    float* o_perm   = o_batch + (size_t)GG * KK;             // [G*K]
    float* o_score  = o_perm + (size_t)GG * KK;              // [N]

    convert_w<<<1, CC, 0, stream>>>(W, W16);
    score_matvec_wmma<<<NN / 128, 256, 0, stream>>>(x, W16, h);
    init_deg<<<NN / 256, 256, 0, stream>>>(dis);
    accum_deg<<<EE / 256, 256, 0, stream>>>(dst, dis);
    dis_self<<<NN / 256, 256, 0, stream>>>(h, dis, acc);
    agg_edges<<<EE / 256, 256, 0, stream>>>(src, dst, dis, h, acc);
    score_final<<<NN / 256, 256, 0, stream>>>(acc, b, o_score);
    topk_kernel<<<GG, 256, 0, stream>>>(o_score, perm, nid, o_perm, o_batch);
    gather_scale<<<GG * KK, 64, 0, stream>>>(x, o_score, perm, o_xnew);
    remap_edges<<<EE / 256, 256, 0, stream>>>(src, dst, nid, o_ei, o_keep);
}
